// NGU_31851477467774
// MI455X (gfx1250) — matrix operand order repaired and checked by reading, not compiled, not run on hardware
//
#include <hip/hip_runtime.h>
#include <math.h>

// ---------- types ----------
typedef __bf16 bf16_t;
typedef __attribute__((ext_vector_type(16))) __bf16 v16bf;
typedef __attribute__((ext_vector_type(8)))  float  v8f;
typedef __attribute__((ext_vector_type(4)))  unsigned v4u;
typedef __attribute__((ext_vector_type(8)))  unsigned v8u;

union Frag { v16bf v; uint4 q[2]; };

#define NROWS 8192
#define KTOP  11      // K+1 smallest (incl. self)
#define CSTRIDE 136   // padded LDS row stride (bf16 elems): 272B = 68 banks -> conflict-free

// ---------- helpers ----------
__device__ inline unsigned short f2bf(float f) {
    union { float f; unsigned u; } x; x.f = f;
    unsigned u = x.u;
    unsigned r = (u + 0x7fffu + ((u >> 16) & 1u)) >> 16;   // RNE
    return (unsigned short)r;
}
__device__ inline float bf2f(unsigned short h) {
    union { unsigned u; float f; } x; x.u = ((unsigned)h) << 16;
    return x.f;
}

// A fragment (16x32 bf16 tile, rows mr..mr+15, k-chunk kb), contiguous k order: k = kb + 16*half + s.
__device__ inline v16bf load_a_frag(const unsigned short* __restrict__ A, int lda,
                                    int mr, int kb, int l15, int half) {
    Frag a;
    const unsigned short* p = A + (size_t)(mr + l15) * lda + kb + half * 16;
    a.q[0] = *(const uint4*)p;          // k = kb+half*16 .. +7
    a.q[1] = *(const uint4*)(p + 8);    // k = kb+half*16+8 .. +15
    return a.v;
}
// B fragment from global: B = W^T (32x16); lane's column of B == contiguous row of W.
__device__ inline v16bf load_b_frag(const unsigned short* __restrict__ W, int ldw,
                                    int nc, int kb, int l15, int half) {
    Frag b;
    const unsigned short* p = W + (size_t)(nc + l15) * ldw + kb + (half << 4);
    b.q[0] = *(const uint4*)p;
    b.q[1] = *(const uint4*)(p + 8);
    return b.v;
}
// B fragment from an LDS-staged 16x128 tile with padded row stride (ds_load_b128 x2).
__device__ inline v16bf load_b_lds(const unsigned short* buf, int kb, int l15, int half) {
    Frag b;
    const unsigned short* p = buf + l15 * CSTRIDE + kb + (half << 4);
    b.q[0] = *(const uint4*)p;
    b.q[1] = *(const uint4*)(p + 8);
    return b.v;
}

// Issue a TDM load: 16 rows x 128 bf16 cols of Z (row-major) -> LDS with pad to 272B stride.
__device__ inline void tdm_issue_tile(const unsigned short* gptr, unsigned lds_off) {
    unsigned long long ga = (unsigned long long)(uintptr_t)gptr;
    v4u g0;
    g0.x = 1u;                                              // count=1, user mode
    g0.y = lds_off;                                         // LDS byte address
    g0.z = (unsigned)ga;                                    // global_addr[31:0]
    g0.w = ((unsigned)(ga >> 32) & 0x01FFFFFFu) | 0x80000000u;  // addr[56:32] | type=2
    v8u g1;
    g1.s0 = (1u << 16)        // data_size = 2 bytes
          | (1u << 20)        // pad_enable
          | (5u << 22)        // pad_interval: every 64 DWORDs (256B row)
          | (3u << 25);       // pad_amount: 4 DWORDs (16B) -> LDS stride 272B
    g1.s1 = (128u << 16);     // tensor_dim0[15:0]=128 in bits[31:16]
    g1.s2 = (8192u << 16);    // tensor_dim0 hi=0 ; tensor_dim1[15:0]=8192
    g1.s3 = (128u << 16);     // tensor_dim1 hi=0 ; tile_dim0=128
    g1.s4 = 16u;              // tile_dim1=16, tile_dim2=0
    g1.s5 = 128u;             // tensor_dim0_stride lo32 = 128 elems
    g1.s6 = 0u;
    g1.s7 = 0u;
    asm volatile("tensor_load_to_lds %0, %1" :: "s"(g0), "s"(g1) : "memory");
}

// ---------- kernels ----------
__global__ void k_convert(const float* __restrict__ in, unsigned short* __restrict__ out, int n) {
    for (int i = blockIdx.x * blockDim.x + threadIdx.x; i < n; i += gridDim.x * blockDim.x)
        out[i] = f2bf(in[i]);
}

// out = [relu](A[M,K] @ W[N,K]^T + bias), bf16 in/out, fp32 accumulate via WMMA.
__global__ __launch_bounds__(128)
void k_gemm(const unsigned short* __restrict__ A, const unsigned short* __restrict__ W,
            const float* __restrict__ bias, unsigned short* __restrict__ out,
            int M, int K, int Nout, int relu) {
    const int lane = threadIdx.x & 31;
    const int wave = threadIdx.x >> 5;
    const int l15  = lane & 15;
    const int half = lane >> 4;
    const int tiles_n = Nout >> 4;
    const int tile = blockIdx.x * 4 + wave;
    if (tile >= (M >> 4) * tiles_n) return;          // wave-uniform: EXEC stays all-ones
    const int mr = (tile / tiles_n) << 4;
    const int nc = (tile % tiles_n) << 4;

    v8f acc = {0.f, 0.f, 0.f, 0.f, 0.f, 0.f, 0.f, 0.f};
    for (int kb = 0; kb < K; kb += 32) {
        v16bf a = load_a_frag(A, K, mr, kb, l15, half);
        v16bf b = load_b_frag(W, K, nc, kb, l15, half);
        acc = __builtin_amdgcn_wmma_f32_16x16x32_bf16(false, a, false, b,
                                                      (short)0, acc, false, false);
    }
    const float bv = bias[nc + l15];
#pragma unroll
    for (int v = 0; v < 8; ++v) {
        float x = acc[v] + bv;
        if (relu) x = x > 0.f ? x : 0.f;
        const int row = mr + half * 8 + v;           // C layout: VGPR v -> M = 8*half+v
        out[(size_t)row * Nout + nc + l15] = f2bf(x);
    }
}

__global__ void k_sq(const unsigned short* __restrict__ Z, float* __restrict__ sq) {
    int i = blockIdx.x * blockDim.x + threadIdx.x;
    if (i >= NROWS) return;
    float s = 0.f;
    for (int k = 0; k < 128; ++k) { float v = bf2f(Z[(size_t)i * 128 + k]); s += v * v; }
    sq[i] = s;
}

__global__ void k_alpha(const unsigned short* __restrict__ h2p, const unsigned short* __restrict__ h2t,
                        const float* __restrict__ w3, const float* __restrict__ b3,
                        const float* __restrict__ wt3, const float* __restrict__ bt3,
                        float* __restrict__ alpha) {
    int i = blockIdx.x * blockDim.x + threadIdx.x;
    if (i >= NROWS) return;
    float p = b3[0], t = bt3[0];
    for (int k = 0; k < 64; ++k) {
        p += bf2f(h2p[(size_t)i * 64 + k]) * w3[k];
        t += bf2f(h2t[(size_t)i * 64 + k]) * wt3[k];
    }
    float d = p - t;
    alpha[i] = d * d;
}

// Fused pairwise-distance + 11-smallest per row.
// 4 waves/block, 32 rows/wave. Column tiles of z staged into LDS by the Tensor Data
// Mover (double-buffered, issued by wave 0, s_wait_tensorcnt + barrier handoff).
__global__ __launch_bounds__(128)
void k_topk(const unsigned short* __restrict__ Z, const float* __restrict__ sq,
            float* __restrict__ sumk) {
    __shared__ float s_sq[NROWS];                        // 32 KB: all sq staged once
    __shared__ float s_d[4][32 * 17];                    // per-wave dist tile, stride 17
    __shared__ __align__(16) unsigned short s_col[2][16 * CSTRIDE];  // TDM double buffer

    for (int i = threadIdx.x; i < NROWS; i += 128) s_sq[i] = sq[i];

    const int lane = threadIdx.x & 31;
    const int wave = threadIdx.x >> 5;
    const int l15  = lane & 15;
    const int half = lane >> 4;
    const int rowbase = (blockIdx.x * 4 + wave) * 32;

    const unsigned lds_col0 = (unsigned)(uintptr_t)&s_col[0][0];
    const unsigned lds_col1 = (unsigned)(uintptr_t)&s_col[1][0];
    if (wave == 0) tdm_issue_tile(Z, lds_col0);          // prefetch tile 0

    // Resident A fragments: rows rowbase..+15 and +16..+31, K=128 in 4 chunks (64 VGPRs)
    v16bf a0[4], a1[4];
#pragma unroll
    for (int kc = 0; kc < 4; ++kc) {
        a0[kc] = load_a_frag(Z, 128, rowbase,      kc * 32, l15, half);
        a1[kc] = load_a_frag(Z, 128, rowbase + 16, kc * 32, l15, half);
    }

    const int myrow = rowbase + lane;                    // each lane owns one full row
    const float sqr = sq[myrow];
    float list[KTOP];
#pragma unroll
    for (int t = 0; t < KTOP; ++t) list[t] = 3.0e38f;

    float* ld = s_d[wave];
    for (int ct = 0; ct < NROWS / 16; ++ct) {
        if (wave == 0) {
            if (ct < NROWS / 16 - 1) {
                // overlap: issue DMA for next tile, then wait for current tile only
                tdm_issue_tile(Z + (size_t)(ct + 1) * 16 * 128,
                               ((ct + 1) & 1) ? lds_col1 : lds_col0);
                __builtin_amdgcn_s_wait_tensorcnt(1);
            } else {
                __builtin_amdgcn_s_wait_tensorcnt(0);
            }
        }
        __syncthreads();                                 // tile ct visible to all waves
        const unsigned short* colbuf = (ct & 1) ? &s_col[1][0] : &s_col[0][0];
        const int colbase = ct << 4;

        v8f c0 = {0.f, 0.f, 0.f, 0.f, 0.f, 0.f, 0.f, 0.f};
        v8f c1 = {0.f, 0.f, 0.f, 0.f, 0.f, 0.f, 0.f, 0.f};
#pragma unroll
        for (int kc = 0; kc < 4; ++kc) {
            v16bf b = load_b_lds(colbuf, kc * 32, l15, half);
            c0 = __builtin_amdgcn_wmma_f32_16x16x32_bf16(false, a0[kc], false, b,
                                                         (short)0, c0, false, false);
            c1 = __builtin_amdgcn_wmma_f32_16x16x32_bf16(false, a1[kc], false, b,
                                                         (short)0, c1, false, false);
        }
        // transpose tile through wave-private LDS so each lane sees its own row
#pragma unroll
        for (int v = 0; v < 8; ++v) {
            ld[(half * 8 + v) * 17 + l15]      = c0[v];
            ld[(16 + half * 8 + v) * 17 + l15] = c1[v];
        }
#pragma unroll
        for (int j = 0; j < 16; ++j) {
            float dot = ld[lane * 17 + j];
            float raw = sqr + s_sq[colbase + j] - 2.0f * dot;
            float d = raw > 0.f ? sqrtf(raw) : 0.f;
            if (d < list[KTOP - 1]) {                    // usually false: exec-masked skip
                float cur = d;
#pragma unroll
                for (int t = 0; t < KTOP; ++t) {
                    float old = list[t];
                    bool sw = cur < old;
                    list[t] = sw ? cur : old;
                    cur     = sw ? old : cur;
                }
            }
        }
        __syncthreads();                                 // all done before buffer reuse
    }
    float tot = 0.f;
#pragma unroll
    for (int t = 0; t < KTOP; ++t) tot += list[t];
    sumk[myrow] = tot - list[0];                         // exclude self (smallest)
}

__global__ void k_reduce(const float* __restrict__ sumk, float* __restrict__ scal) {
    __shared__ float red[256];
    float s = 0.f;
    for (int i = threadIdx.x; i < NROWS; i += 256) s += sumk[i];
    red[threadIdx.x] = s;
    __syncthreads();
    for (int off = 128; off > 0; off >>= 1) {
        if ((int)threadIdx.x < off) red[threadIdx.x] += red[threadIdx.x + off];
        __syncthreads();
    }
    if (threadIdx.x == 0) { float m = red[0] / (float)NROWS; scal[0] = m * m; }
}

__global__ void k_final(const float* __restrict__ sumk, const float* __restrict__ alpha,
                        const float* __restrict__ scal, float* __restrict__ out) {
    int i = blockIdx.x * blockDim.x + threadIdx.x;
    if (i >= NROWS) return;
    const float meansq = scal[0];
    float sk = sumk[i];
    float knn = 0.001f / (sk * sk / meansq + 0.001f);
    float rep = 1.0f / (sqrtf(knn) + 0.001f);
    float a = alpha[i];
    a = a < 1.0f ? 1.0f : (a > 5.0f ? 5.0f : a);
    out[i] = rep * a;
}

// ---------- host ----------
extern "C" void kernel_launch(void* const* d_in, const int* in_sizes, int n_in,
                              void* d_out, int out_size, void* d_ws, size_t ws_size,
                              hipStream_t stream) {
    const float* s   = (const float*)d_in[0];
    const float* w1  = (const float*)d_in[1];  const float* b1  = (const float*)d_in[2];
    const float* w2  = (const float*)d_in[3];  const float* b2  = (const float*)d_in[4];
    const float* w3  = (const float*)d_in[5];  const float* b3  = (const float*)d_in[6];
    const float* wt1 = (const float*)d_in[7];  const float* bt1 = (const float*)d_in[8];
    const float* wt2 = (const float*)d_in[9];  const float* bt2 = (const float*)d_in[10];
    const float* wt3 = (const float*)d_in[11]; const float* bt3 = (const float*)d_in[12];
    const float* wz1 = (const float*)d_in[13]; const float* bz1 = (const float*)d_in[14];
    const float* wz2 = (const float*)d_in[15]; const float* bz2 = (const float*)d_in[16];
    const float* wz3 = (const float*)d_in[17]; const float* bz3 = (const float*)d_in[18];

    char* ws = (char*)d_ws;
    size_t off = 0;
    auto take = [&](size_t bytes) -> char* {
        char* p = ws + off;
        off = (off + bytes + 255) & ~(size_t)255;
        return p;
    };
    unsigned short* sbf   = (unsigned short*)take((size_t)NROWS * 512 * 2);
    unsigned short* w1b   = (unsigned short*)take(128 * 512 * 2);
    unsigned short* wt1b  = (unsigned short*)take(128 * 512 * 2);
    unsigned short* wz1b  = (unsigned short*)take(128 * 512 * 2);
    unsigned short* w2b   = (unsigned short*)take(64 * 128 * 2);
    unsigned short* wt2b  = (unsigned short*)take(64 * 128 * 2);
    unsigned short* wz2b  = (unsigned short*)take(64 * 128 * 2);
    unsigned short* wz3b  = (unsigned short*)take(128 * 64 * 2);
    unsigned short* h1p   = (unsigned short*)take((size_t)NROWS * 128 * 2);
    unsigned short* h1t   = (unsigned short*)take((size_t)NROWS * 128 * 2);
    unsigned short* h1z   = (unsigned short*)take((size_t)NROWS * 128 * 2);
    unsigned short* h2p   = (unsigned short*)take((size_t)NROWS * 64 * 2);
    unsigned short* h2t   = (unsigned short*)take((size_t)NROWS * 64 * 2);
    unsigned short* h2z   = (unsigned short*)take((size_t)NROWS * 64 * 2);
    unsigned short* zbf   = (unsigned short*)take((size_t)NROWS * 128 * 2);
    float* sq    = (float*)take(NROWS * 4);
    float* alpha = (float*)take(NROWS * 4);
    float* sumk  = (float*)take(NROWS * 4);
    float* scal  = (float*)take(256);

    // fp32 -> bf16 conversions
    k_convert<<<4096, 256, 0, stream>>>(s,   sbf,  NROWS * 512);
    k_convert<<<256,  256, 0, stream>>>(w1,  w1b,  128 * 512);
    k_convert<<<256,  256, 0, stream>>>(wt1, wt1b, 128 * 512);
    k_convert<<<256,  256, 0, stream>>>(wz1, wz1b, 128 * 512);
    k_convert<<<32,   256, 0, stream>>>(w2,  w2b,  64 * 128);
    k_convert<<<32,   256, 0, stream>>>(wt2, wt2b, 64 * 128);
    k_convert<<<32,   256, 0, stream>>>(wz2, wz2b, 64 * 128);
    k_convert<<<32,   256, 0, stream>>>(wz3, wz3b, 128 * 64);

    // layer 1: [8192,512] @ [512,128]^T  -> 4096 tiles, 4 waves/block
    k_gemm<<<1024, 128, 0, stream>>>(sbf, w1b,  b1,  h1p, NROWS, 512, 128, 1);
    k_gemm<<<1024, 128, 0, stream>>>(sbf, wt1b, bt1, h1t, NROWS, 512, 128, 1);
    k_gemm<<<1024, 128, 0, stream>>>(sbf, wz1b, bz1, h1z, NROWS, 512, 128, 1);
    // layer 2: [8192,128] @ [128,64]^T
    k_gemm<<<512,  128, 0, stream>>>(h1p, w2b,  b2,  h2p, NROWS, 128, 64, 1);
    k_gemm<<<512,  128, 0, stream>>>(h1t, wt2b, bt2, h2t, NROWS, 128, 64, 1);
    k_gemm<<<512,  128, 0, stream>>>(h1z, wz2b, bz2, h2z, NROWS, 128, 64, 1);
    // z layer 3: [8192,64] @ [64,128]^T (no relu)
    k_gemm<<<1024, 128, 0, stream>>>(h2z, wz3b, bz3, zbf, NROWS, 64, 128, 0);

    k_sq<<<32, 256, 0, stream>>>(zbf, sq);
    k_alpha<<<32, 256, 0, stream>>>(h2p, h2t, w3, b3, wt3, bt3, alpha);

    // fused NxN distance + top-11 (dominant kernel): 64 blocks x 128 threads,
    // TDM-staged column tiles, double buffered
    k_topk<<<64, 128, 0, stream>>>(zbf, sq, sumk);

    k_reduce<<<1, 256, 0, stream>>>(sumk, scal);
    k_final<<<32, 256, 0, stream>>>(sumk, alpha, scal, (float*)d_out);
}
